// reaction_diffusion_gcn_90872918049148
// MI455X (gfx1250) — compile-verified, matrix operand order checked by reading, not run on hardware
//
#include <hip/hip_runtime.h>
#include <hip/hip_bf16.h>

// CDNA5 / gfx1250 implementation.
//
// Plan:
//  kernel 1 (build_mats): for each of the 288 slots, materialize dense
//    R = A_r + diag(colsum(A_r)) and D = diag(colsum(A_d)) - A_d as f16,
//    padded to [208 rows][224 cols] (K multiple of 32) in d_ws.
//    Workspace requirement: 288 * 2 * 208 * 224 * 2 B = ~53.7 MB.
//  kernel 2 (rd_gcn): one workgroup (256 thr = 8 wave32) per batch sample;
//    dense GEMMs R@x and D@x via v_wmma_f32_16x16x32_f16 with f32 accum,
//    B fragment shared between the R and D WMMAs; fused tanh/bias/self-loop
//    epilogue.

typedef _Float16 half_t;
typedef __attribute__((ext_vector_type(16))) _Float16 v16h;
typedef __attribute__((ext_vector_type(8)))  _Float16 v8h;
typedef __attribute__((ext_vector_type(8)))  float    v8f;

#define N_NODES 207
#define N_EDGES 1722
#define N_SLOTS 288
#define LFEAT   24
#define MPAD    208                 // padded rows (13 * 16)
#define KPAD    224                 // padded cols (7 * 32)
#define MAT_ELEMS  (MPAD * KPAD)    // halves per matrix
#define SLOT_ELEMS (2 * MAT_ELEMS)  // [R | D] per slot

// ---------------------------------------------------------------------------
// Kernel 1: build per-slot dense f16 operators. One block per slot, thread w
// owns row w of both R and D (no atomics; duplicate edges accumulate via
// sequential same-thread RMW, which the memory system keeps in order).
// ---------------------------------------------------------------------------
__global__ __launch_bounds__(256)
void build_mats(const float* __restrict__ wr, const float* __restrict__ wd,
                const int* __restrict__ edge_index, half_t* __restrict__ mats) {
  __shared__ int   ea[N_EDGES];
  __shared__ int   eb[N_EDGES];
  __shared__ float lwr[N_EDGES];
  __shared__ float lwd[N_EDGES];

  const int s   = blockIdx.x;
  const int tid = threadIdx.x;
  half_t* slot = mats + (size_t)s * SLOT_ELEMS;

  // Stage edge list + this slot's edge weights in LDS.
  for (int e = tid; e < N_EDGES; e += 256) {
    ea[e]  = edge_index[e];
    eb[e]  = edge_index[N_EDGES + e];
    lwr[e] = wr[(size_t)s * N_EDGES + e];
    lwd[e] = wd[(size_t)s * N_EDGES + e];
  }
  // Zero this slot's R|D region (coalesced; also provides the zero padding
  // rows/cols the GEMM relies on).
  for (int i = tid; i < SLOT_ELEMS; i += 256) slot[i] = (half_t)0.f;
  __syncthreads();

  const int w = tid;
  if (w < N_NODES) {
    half_t* Rrow = slot + (size_t)w * KPAD;
    half_t* Drow = slot + MAT_ELEMS + (size_t)w * KPAD;
    float degr = 0.f, degd = 0.f;
    for (int e = 0; e < N_EDGES; ++e) {
      const int a = ea[e], c = eb[e];
      if (a == w || c == w) {
        const float vr = lwr[e], vd = lwd[e];
        if (a == w) {  // A[w][c] += v ; d[w] += v
          Rrow[c] = (half_t)((float)Rrow[c] + vr);
          Drow[c] = (half_t)((float)Drow[c] - vd);
          degr += vr; degd += vd;
        }
        if (c == w) {  // A[w][a] += v ; d[w] += v
          Rrow[a] = (half_t)((float)Rrow[a] + vr);
          Drow[a] = (half_t)((float)Drow[a] - vd);
          degr += vr; degd += vd;
        }
      }
    }
    // R = A + diag(d) ; D = diag(d) - A
    Rrow[w] = (half_t)((float)Rrow[w] + degr);
    Drow[w] = (half_t)((float)Drow[w] + degd);
  }
}

// ---------------------------------------------------------------------------
// Kernel 2: per-batch fused GEMM + epilogue. 8 wave32s; each wave owns a
// 16x16 output tile pair (R-accum, D-accum), K-loop of 7 steps of K=32.
// ---------------------------------------------------------------------------
__global__ __launch_bounds__(256)
void rd_gcn(const float* __restrict__ x_in,    // [B, 24, 207] (= [B,2,12,N])
            const float* __restrict__ bias_r,  // [S, 207]
            const float* __restrict__ bias_d,  // [S, 207]
            const float* __restrict__ wsl,     // [S, 207]
            const int*   __restrict__ ind,     // [B]
            const half_t* __restrict__ mats,   // [S][R|D][208][224] f16
            float* __restrict__ out) {         // [B, 207, 24]
  __shared__ half_t x_h[32 * KPAD];     // B-matrix staging: [l][v] f16
  __shared__ float  x_f[MPAD * 25];     // epilogue x:      [v][l] f32
  __shared__ float  br_l[MPAD];
  __shared__ float  bd_l[MPAD];
  __shared__ float  wsl_l[MPAD];

  const int b   = blockIdx.x;
  const int tid = threadIdx.x;
  const int s   = ind[b];

  for (int i = tid; i < 32 * KPAD; i += 256) x_h[i] = (half_t)0.f;
  __syncthreads();

  const float* xb = x_in + (size_t)b * (LFEAT * N_NODES);
  for (int i = tid; i < LFEAT * N_NODES; i += 256) {
    const int l = i / N_NODES;
    const int v = i - l * N_NODES;      // x[v][l] = inputs[b][l][v]
    const float val = xb[i];
    x_f[v * 25 + l]   = val;
    x_h[l * KPAD + v] = (half_t)val;
  }
  for (int i = tid; i < N_NODES; i += 256) {
    br_l[i]  = bias_r[(size_t)s * N_NODES + i];
    bd_l[i]  = bias_d[(size_t)s * N_NODES + i];
    wsl_l[i] = wsl  [(size_t)s * N_NODES + i];
  }
  __syncthreads();

  const half_t* Rmat = mats + (size_t)s * SLOT_ELEMS;
  const half_t* Dmat = Rmat + MAT_ELEMS;

  const int lane  = tid & 31;
  const int wid   = tid >> 5;
  const int nlow  = lane & 15;
  const int khalf = lane >> 4;

  // 13 M-tiles x 2 N-tiles = 26 tile pairs over 8 waves.
  for (int t = wid; t < 26; t += 8) {
    const int tm = t % 13, tn = t / 13;
    const int m0 = tm * 16, n0 = tn * 16;
    v8f cr = {};
    v8f cd = {};
    const int arow = m0 + nlow;   // A row this lane carries (lane & lane+16)
    for (int k0 = 0; k0 < KPAD; k0 += 32) {
      const int acol = k0 + khalf * 8;
      // A fragments (ISA 16-bit A 16x32 layout): halves 0-7 = K acol..acol+7,
      // halves 8-15 = K acol+16..acol+23.
      const v8h rl = *(const v8h*)(Rmat + (size_t)arow * KPAD + acol);
      const v8h rh = *(const v8h*)(Rmat + (size_t)arow * KPAD + acol + 16);
      const v8h dl = *(const v8h*)(Dmat + (size_t)arow * KPAD + acol);
      const v8h dh = *(const v8h*)(Dmat + (size_t)arow * KPAD + acol + 16);
      // B fragment (32x16, column n0+nlow) from LDS: two ds_load_b128s.
      const int brow = n0 + nlow;
      const v8h bl = *(const v8h*)(&x_h[brow * KPAD + acol]);
      const v8h bh = *(const v8h*)(&x_h[brow * KPAD + acol + 16]);
      v16h ar, ad, bb;
#pragma unroll
      for (int i = 0; i < 8; ++i) {
        ar[i] = rl[i]; ar[i + 8] = rh[i];
        ad[i] = dl[i]; ad[i + 8] = dh[i];
        bb[i] = bl[i]; bb[i + 8] = bh[i];
      }
      cr = __builtin_amdgcn_wmma_f32_16x16x32_f16(false, ar, false, bb,
                                                  (short)0, cr, false, false);
      cd = __builtin_amdgcn_wmma_f32_16x16x32_f16(false, ad, false, bb,
                                                  (short)0, cd, false, false);
    }
    // Epilogue. C layout: vgpr r, lane -> row m0+r(+8 for lanes 16-31),
    // col n0 + lane%16.
    const int l = n0 + nlow;
#pragma unroll
    for (int r = 0; r < 8; ++r) {
      const int row = m0 + r + (khalf ? 8 : 0);
      if (row < N_NODES && l < LFEAT) {
        const float xv  = x_f[row * 25 + l];
        const float rea = tanhf(cr[r] + br_l[row]);
        const float res = rea + cd[r] + bd_l[row] + xv * (1.f + wsl_l[row]);
        out[(size_t)b * (N_NODES * LFEAT) + row * 24 + l] = res;
      }
    }
  }
}

// ---------------------------------------------------------------------------
extern "C" void kernel_launch(void* const* d_in, const int* in_sizes, int n_in,
                              void* d_out, int out_size, void* d_ws, size_t ws_size,
                              hipStream_t stream) {
  const float* inputs = (const float*)d_in[0];
  const float* wr     = (const float*)d_in[1];
  const float* wd     = (const float*)d_in[2];
  const float* br     = (const float*)d_in[3];
  const float* bd     = (const float*)d_in[4];
  const float* wsl    = (const float*)d_in[5];
  const int*   ind    = (const int*)d_in[6];
  const int*   ei     = (const int*)d_in[7];
  float* out   = (float*)d_out;
  half_t* mats = (half_t*)d_ws;   // needs ~53.7 MB of workspace

  const int B = in_sizes[0] / (2 * 12 * N_NODES);

  build_mats<<<N_SLOTS, 256, 0, stream>>>(wr, wd, ei, mats);
  rd_gcn<<<B, 256, 0, stream>>>(inputs, br, bd, wsl, ind, mats, out);
}
